// MultiHeadEdgeToNodeAttention_73718818669321
// MI455X (gfx1250) — matrix operand order repaired and checked by reading, not compiled, not run on hardware
//
#include <hip/hip_runtime.h>

#define N_NODES 32768
#define N_EDGES 8192
#define ND 128
#define HD 64
#define OD 128
#define NH 4
#define EPSV 1e-8f

typedef __attribute__((ext_vector_type(16))) __bf16   v16bf;
typedef __attribute__((ext_vector_type(8)))  float    v8f;
typedef __attribute__((ext_vector_type(4)))  unsigned u32x4;
typedef __attribute__((ext_vector_type(4)))  float    f32x4;

union FragB { v16bf bf; u32x4 q[2]; };
union BfBits { __bf16 b; unsigned short s; };

__device__ __forceinline__ __bf16 bfc(float f) { return (__bf16)f; }
__device__ __forceinline__ unsigned short bfbits(float f) { BfBits u; u.b = (__bf16)f; return u.s; }

// monotonic float <-> sortable-uint mapping for integer atomic min/max
__device__ __forceinline__ unsigned f2s(float f) {
    unsigned u = __float_as_uint(f);
    return (u & 0x80000000u) ? ~u : (u | 0x80000000u);
}
__device__ __forceinline__ float s2f(unsigned s) {
    unsigned u = (s & 0x80000000u) ? (s & 0x7fffffffu) : ~s;
    return __uint_as_float(u);
}

__device__ __forceinline__ v8f vzero() {
    v8f z = {0.f, 0.f, 0.f, 0.f, 0.f, 0.f, 0.f, 0.f};
    return z;
}

__device__ __forceinline__ v8f wmma_bf16(v16bf a, v16bf b, v8f c) {
    return __builtin_amdgcn_wmma_f32_16x16x32_bf16(false, a, false, b, (short)0, c, false, false);
}

// A fragment (16x32, M x K) from row-major fp32: p points at (row=lane&15, k0 + (lane>>4)*8)
// lane<16: K = {0..7, 16..23}; lane>=16: K = {8..15, 24..31}  (ISA 7.12.2)
__device__ __forceinline__ v16bf load_a_g32(const float* p) {
    v16bf a;
#pragma unroll
    for (int i = 0; i < 8; i++) a[i] = bfc(p[i]);
#pragma unroll
    for (int i = 0; i < 8; i++) a[8 + i] = bfc(p[16 + i]);
    return a;
}

// A fragment from row-major bf16 (LDS): rowbase points at (row=lane&15, k0 + (lane>>4)*8)
__device__ __forceinline__ v16bf load_a_lds_bf(const __bf16* rowbase) {
    FragB f;
    f.q[0] = *((const u32x4*)rowbase);        // K +0..7
    f.q[1] = *((const u32x4*)(rowbase + 16)); // K +16..23
    return f.bf;
}

// B fragment (32x16, K x N) from transposed bf16 buffer laid out [n][k] with row stride kstride.
// lane<16 holds K={0..7,16..23} for col n=lane; lane>=16 holds K={8..15,24..31}.
__device__ __forceinline__ v16bf load_b_lds(const __bf16* base, int kstride, int k0, int lane) {
    int n = lane & 15, kh = lane >> 4;
    const __bf16* p = base + n * kstride + k0 + kh * 8;
    FragB f;
    const u32x4* q = (const u32x4*)p;
    f.q[0] = q[0];
    f.q[1] = q[2];
    return f.bf;
}

// Transpose-stage a row-major fp32 [K][Ncols] weight into LDS as bf16 [Ncols][K].
// Requires K*Ncols == 8192 and Ncols % 32 == 0 (both weight shapes here).
__device__ __forceinline__ void stage_WT(const float* W, __bf16* lds, int K, int Ncols, int tid) {
    int base = tid * 32;                 // 256 threads * 32 elems = 8192
    int k = base / Ncols;
    int n0 = base % Ncols;
#pragma unroll 8
    for (int j = 0; j < 32; j++) lds[(n0 + j) * K + k] = bfc(W[base + j]);
}

// Async global -> LDS copy of 64 bytes (4 x b128); IOFFSET advances both sides.
// Tracked by ASYNCcnt; consumer must s_wait_asynccnt + barrier before reading.
__device__ __forceinline__ void async_copy64(void* lds_dst, const void* gsrc) {
    unsigned ldsaddr = (unsigned)(uintptr_t)lds_dst;      // LDS aperture: addr[31:0] = LDS offset
    unsigned long long gaddr = (unsigned long long)(uintptr_t)gsrc;
    asm volatile(
        "global_load_async_to_lds_b128 %0, %1, off\n\t"
        "global_load_async_to_lds_b128 %0, %1, off offset:16\n\t"
        "global_load_async_to_lds_b128 %0, %1, off offset:32\n\t"
        "global_load_async_to_lds_b128 %0, %1, off offset:48"
        :: "v"(ldsaddr), "v"(gaddr) : "memory");
}
__device__ __forceinline__ void wait_async0() {
    asm volatile("s_wait_asynccnt 0x0" ::: "memory");
}

// ---------------------------------------------------------------------------
// K0: te_h = edge_features @ edgeW[h] + edgeB[h]   (all heads, WMMA bf16)
// Output stored transposed bf16: teT[(h*64+n)*8192 + e]
// ---------------------------------------------------------------------------
__global__ void __launch_bounds__(256) te_kernel(const float* __restrict__ edge_features,
                                                 const float* __restrict__ edgeW,
                                                 const float* __restrict__ edgeB,
                                                 __bf16* __restrict__ teT) {
    __shared__ __bf16 ldsW[HD * ND]; // [64 n][128 k]
    int h = blockIdx.x >> 6;         // 64 blocks per head
    stage_WT(edgeW + (size_t)h * ND * HD, ldsW, ND, HD, threadIdx.x);
    __syncthreads();

    int wave = threadIdx.x >> 5, lane = threadIdx.x & 31;
    int ln = lane & 15, g = lane >> 4, mo = g * 8;
    int row0 = ((blockIdx.x & 63) * 8 + wave) * 16;

    v8f acc[4];
#pragma unroll
    for (int nt = 0; nt < 4; nt++) acc[nt] = vzero();

#pragma unroll
    for (int kk = 0; kk < 4; kk++) {
        const float* ap = edge_features + (size_t)(row0 + ln) * ND + kk * 32 + g * 8;
        v16bf a = load_a_g32(ap);
#pragma unroll
        for (int nt = 0; nt < 4; nt++) {
            v16bf b = load_b_lds(ldsW + nt * 16 * ND, ND, kk * 32, lane);
            acc[nt] = wmma_bf16(a, b, acc[nt]);
        }
    }
#pragma unroll
    for (int nt = 0; nt < 4; nt++) {
        float bias = edgeB[h * HD + nt * 16 + ln];
        u32x4 pk;
#pragma unroll
        for (int i = 0; i < 4; i++) {
            pk[i] = (unsigned)bfbits(acc[nt][2 * i] + bias) |
                    ((unsigned)bfbits(acc[nt][2 * i + 1] + bias) << 16);
        }
        *(u32x4*)(teT + (size_t)(h * HD + nt * 16 + ln) * N_EDGES + row0 + mo) = pk;
    }
}

// ---------------------------------------------------------------------------
// K1: agg[h] = (incidence @ te_h) / deg  for ALL 4 heads in ONE incidence pass.
// teT tiles staged via double-buffered GLOBAL_LOAD_ASYNC_TO_LDS_B128 (ASYNCcnt),
// overlapping the copy of K-chunk i+1 with the 16 WMMAs of chunk i.
// deg (row sums + EPS) fused from the A-fragment fp32 values.
// ---------------------------------------------------------------------------
__global__ void __launch_bounds__(256) agg_kernel(const float* __restrict__ inc,
                                                  const __bf16* __restrict__ teT,
                                                  float* __restrict__ agg) {
    __shared__ __bf16 ldsB[2][256 * 32]; // double-buffered: 2 x 16KB
    int tid = threadIdx.x, wave = tid >> 5, lane = tid & 31;
    int ln = lane & 15, g = lane >> 4;
    int m0 = blockIdx.x * 128 + wave * 16;

    v8f acc[16];
#pragma unroll
    for (int s = 0; s < 16; s++) acc[s] = vzero();
    float rs = 0.f;

    const __bf16* grow = teT + (size_t)tid * N_EDGES; // this thread's (h,n) row

    // prologue: stage chunk 0
    async_copy64(&ldsB[0][tid * 32], grow);
    wait_async0();
    __syncthreads();

    int cur = 0;
    for (int k0 = 0; k0 < N_EDGES; k0 += 32) {
        // prefetch next chunk into the other buffer (overlaps with WMMAs below)
        if (k0 + 32 < N_EDGES)
            async_copy64(&ldsB[cur ^ 1][tid * 32], grow + k0 + 32);

        const float* p = inc + (size_t)(m0 + ln) * N_EDGES + k0 + g * 8;
        v16bf a;
#pragma unroll
        for (int i = 0; i < 8; i++) { float f = p[i];      rs += f; a[i] = bfc(f); }
#pragma unroll
        for (int i = 0; i < 8; i++) { float f = p[16 + i]; rs += f; a[8 + i] = bfc(f); }

        const __bf16* bb = ldsB[cur];
#pragma unroll
        for (int s = 0; s < 16; s++) { // s = h*4 + nt
            v16bf b = load_b_lds(bb + s * 16 * 32, 32, 0, lane);
            acc[s] = wmma_bf16(a, b, acc[s]);
        }

        wait_async0();   // next buffer's async stores (this wave's share) landed
        __syncthreads(); // all waves done staging + done reading cur
        cur ^= 1;
    }

    // full row sum for row (lane&15): combine the two K-halves held by the lane pair
    rs += __shfl_xor(rs, 16, 32);
    float inv[8];
#pragma unroll
    for (int v = 0; v < 8; v++) {
        float d = __shfl(rs, v + g * 8, 32); // deg of row m0 + g*8 + v
        inv[v] = 1.f / (d + EPSV);
    }

#pragma unroll
    for (int h = 0; h < NH; h++) {
#pragma unroll
        for (int nt = 0; nt < 4; nt++) {
            float* out = agg + ((size_t)h * N_NODES + m0) * HD + nt * 16 + ln;
            v8f A = acc[h * 4 + nt];
#pragma unroll
            for (int v = 0; v < 8; v++) out[(size_t)(g * 8 + v) * HD] = A[v] * inv[v];
        }
    }
}

// ---------------------------------------------------------------------------
// K2 (per head): tn GEMM + attention gate + out GEMM, fused. Tracks per-column
// min/max of upd via sortable-uint atomics (LDS first, then global).
// ---------------------------------------------------------------------------
__global__ void __launch_bounds__(256) head_kernel(const float* __restrict__ x_in,
                                                   const float* __restrict__ nodeW,
                                                   const float* __restrict__ nodeB,
                                                   const float* __restrict__ attnW,
                                                   const float* __restrict__ attnB,
                                                   const float* __restrict__ outW,
                                                   const float* __restrict__ outB,
                                                   const float* __restrict__ agg_h,
                                                   float* __restrict__ upd_out,
                                                   unsigned* __restrict__ gmin,
                                                   unsigned* __restrict__ gmax, int h) {
    __shared__ __bf16 ldsNW[HD * ND];   // nodeW^T  [64 n][128 k]
    __shared__ __bf16 ldsOW[OD * HD];   // outW^T   [128 n][64 k]
    __shared__ __bf16 ldsU[8][16 * HD]; // per-wave upd tile [16 m][64 n]
    __shared__ unsigned lmin[OD], lmax[OD];

    int tid = threadIdx.x;
    stage_WT(nodeW + (size_t)h * ND * HD, ldsNW, ND, HD, tid);
    stage_WT(outW + (size_t)h * HD * OD, ldsOW, HD, OD, tid);
    if (tid < OD) { lmin[tid] = 0xffffffffu; lmax[tid] = 0u; }
    __syncthreads();

    int wave = tid >> 5, lane = tid & 31;
    int ln = lane & 15, g = lane >> 4, mo = g * 8;
    int m0 = blockIdx.x * 128 + wave * 16;

    // ---- tn = x @ nodeW[h] ----
    v8f tn[4];
#pragma unroll
    for (int nt = 0; nt < 4; nt++) tn[nt] = vzero();
#pragma unroll
    for (int kk = 0; kk < 4; kk++) {
        const float* p = x_in + (size_t)(m0 + ln) * ND + kk * 32 + g * 8;
        v16bf a = load_a_g32(p);
#pragma unroll
        for (int nt = 0; nt < 4; nt++) {
            v16bf b = load_b_lds(ldsNW + nt * 16 * ND, ND, kk * 32, lane);
            tn[nt] = wmma_bf16(a, b, tn[nt]);
        }
    }

    // ---- bias, agg, attention scores ----
    v8f ag[4];
    float sp[8] = {0.f, 0.f, 0.f, 0.f, 0.f, 0.f, 0.f, 0.f};
#pragma unroll
    for (int nt = 0; nt < 4; nt++) {
        float nb = nodeB[h * HD + nt * 16 + ln];
        float aw = attnW[h * HD + nt * 16 + ln];
        const float* ap = agg_h + (size_t)m0 * HD + nt * 16 + ln;
#pragma unroll
        for (int v = 0; v < 8; v++) {
            tn[nt][v] += nb;
            float av = ap[(size_t)(mo + v) * HD];
            ag[nt][v] = av;
            sp[v] += (tn[nt][v] + av) * aw;
        }
    }
    // reduce the 64-wide dot across the 16 lanes of this half-wave
#pragma unroll
    for (int mask = 1; mask <= 8; mask <<= 1) {
#pragma unroll
        for (int v = 0; v < 8; v++) sp[v] += __shfl_xor(sp[v], mask, 32);
    }
    float ab = attnB[h];
    float coeff[8];
#pragma unroll
    for (int v = 0; v < 8; v++) {
        float s = sp[v] + ab;
        s = (s >= 0.f) ? s : 0.2f * s;      // LeakyReLU(0.2)
        coeff[v] = 1.f / (1.f + __expf(-s)); // sigmoid
    }

    // ---- upd = coeff*agg + tn -> per-wave LDS (C-layout -> row-major bf16) ----
    __bf16* u = ldsU[wave];
#pragma unroll
    for (int nt = 0; nt < 4; nt++) {
#pragma unroll
        for (int v = 0; v < 8; v++)
            u[(mo + v) * HD + nt * 16 + ln] = bfc(coeff[v] * ag[nt][v] + tn[nt][v]);
    }
    // same-wave LDS ops are in-order: reads below see the writes above

    // ---- out = upd @ outW[h] + outB[h] ----
    v8f oc[8];
#pragma unroll
    for (int s = 0; s < 8; s++) oc[s] = vzero();
#pragma unroll
    for (int kk = 0; kk < 2; kk++) {
        const __bf16* ar = u + ln * HD + kk * 32 + g * 8;
        v16bf a = load_a_lds_bf(ar);
#pragma unroll
        for (int s = 0; s < 8; s++) {
            v16bf b = load_b_lds(ldsOW + s * 16 * HD, HD, kk * 32, lane);
            oc[s] = wmma_bf16(a, b, oc[s]);
        }
    }
#pragma unroll
    for (int s = 0; s < 8; s++) {
        float ob = outB[h * OD + s * 16 + ln];
        float* op = upd_out + (size_t)m0 * OD + s * 16 + ln;
        float mn = 3.4e38f, mx = -3.4e38f;
#pragma unroll
        for (int v = 0; v < 8; v++) {
            float val = oc[s][v] + ob;
            op[(size_t)(mo + v) * OD] = val;
            mn = fminf(mn, val);
            mx = fmaxf(mx, val);
        }
        atomicMin(&lmin[s * 16 + ln], f2s(mn));
        atomicMax(&lmax[s * 16 + ln], f2s(mx));
    }
    __syncthreads();
    if (tid < OD) {
        atomicMin(&gmin[tid], lmin[tid]);
        atomicMax(&gmax[tid], lmax[tid]);
    }
}

__global__ void init_minmax(unsigned* __restrict__ gmin, unsigned* __restrict__ gmax) {
    int t = threadIdx.x;
    gmin[t] = 0xffffffffu;
    gmax[t] = 0u;
}

// K3: x = relu((upd - colmin) / (colmax - colmin + EPS))
__global__ void __launch_bounds__(256) norm_kernel(const float* __restrict__ src,
                                                   float* __restrict__ dst,
                                                   const unsigned* __restrict__ cmin,
                                                   const unsigned* __restrict__ cmax) {
    __shared__ float smn[OD], sinv[OD];
    int t = threadIdx.x;
    if (t < OD) {
        float mn = s2f(cmin[t]);
        float mx = s2f(cmax[t]);
        smn[t] = mn;
        sinv[t] = 1.f / (mx - mn + EPSV);
    }
    __syncthreads();
    size_t idx = ((size_t)blockIdx.x * 256 + t) * 4;
    f32x4 v = *(const f32x4*)(src + idx);
    int c0 = (int)(idx & (OD - 1));
    f32x4 r;
#pragma unroll
    for (int j = 0; j < 4; j++) {
        float val = (v[j] - smn[c0 + j]) * sinv[c0 + j];
        r[j] = val > 0.f ? val : 0.f;
    }
    *(f32x4*)(dst + idx) = r;
}

extern "C" void kernel_launch(void* const* d_in, const int* in_sizes, int n_in,
                              void* d_out, int out_size, void* d_ws, size_t ws_size,
                              hipStream_t stream) {
    (void)in_sizes; (void)n_in; (void)out_size; (void)ws_size;
    const float* node_features = (const float*)d_in[0];
    const float* incidence     = (const float*)d_in[1];
    const float* edge_features = (const float*)d_in[2];
    const float* nodeW = (const float*)d_in[3];
    const float* nodeB = (const float*)d_in[4];
    const float* edgeW = (const float*)d_in[5];
    const float* edgeB = (const float*)d_in[6];
    const float* attnW = (const float*)d_in[7];
    const float* attnB = (const float*)d_in[8];
    const float* outW  = (const float*)d_in[9];
    const float* outB  = (const float*)d_in[10];

    char* ws = (char*)d_ws;
    __bf16*   teT  = (__bf16*)ws;                        //  4 MB: [4*64][8192] bf16
    float*    agg  = (float*)(ws + (4ull << 20));        // 32 MB: [4][N][64] f32
    float*    xbuf = (float*)(ws + (36ull << 20));       // 16 MB: [N][128] f32
    unsigned* gmin = (unsigned*)(ws + (52ull << 20));
    unsigned* gmax = gmin + OD;

    te_kernel<<<256, 256, 0, stream>>>(edge_features, edgeW, edgeB, teT);
    agg_kernel<<<256, 256, 0, stream>>>(incidence, teT, agg);

    const float* xin = node_features;
    for (int h = 0; h < NH; h++) {
        init_minmax<<<1, OD, 0, stream>>>(gmin, gmax);
        head_kernel<<<256, 256, 0, stream>>>(xin, nodeW, nodeB, attnW, attnB, outW, outB,
                                             agg + (size_t)h * N_NODES * HD, xbuf,
                                             gmin, gmax, h);
        float* dst = (h == NH - 1) ? (float*)d_out : xbuf;
        norm_kernel<<<4096, 256, 0, stream>>>(xbuf, dst, gmin, gmax);
        xin = xbuf;
    }
}
